// HybridMoELayer_26353919328877
// MI455X (gfx1250) — matrix operand order repaired and checked by reading, not compiled, not run on hardware
//
#include <hip/hip_runtime.h>
#include <hip/hip_bf16.h>

typedef __attribute__((ext_vector_type(16))) __bf16 v16bf;
typedef __attribute__((ext_vector_type(8)))  float  v8f;

#define T_TOK 1024
#define DIM   2880
#define DIM2  5760
#define NEXP  12
#define NN    8
#define NC    4
#define ICOMP 512
#define TOPK  4

#define BM 128
#define BN 64
#define BK 64
#define LDA  (BK + 8)   /* A LDS row stride (elements): 144B/row -> 16B aligned */
#define LDKB (BK + 8)   /* B LDS (transposed, K-contiguous) stride per column  */

union BF16x16 { uint4 q[2]; v16bf v; };

__device__ __forceinline__ float f4_at(const float4& v, int j) {
  return j == 0 ? v.x : j == 1 ? v.y : j == 2 ? v.z : v.w;
}

// Pack two f32 -> one dword of two bf16 with a single hardware instruction.
// (The union idiom lowers to cvt + bit-stitching; force the packed form.)
__device__ __forceinline__ unsigned int pk_bf16(float lo, float hi) {
  unsigned int r;
  asm("v_cvt_pk_bf16_f32 %0, %1, %2" : "=v"(r) : "v"(lo), "v"(hi));
  return r;
}

// ---------------------------------------------------------------------------
// WMMA fragment loaders matching CDNA5 VGPR layouts (CDNA5 ISA)
// ---------------------------------------------------------------------------
__device__ __forceinline__ v16bf load_a_frag(const __bf16* As_kk, int lane) {
  const int m  = lane & 15;
  const int kb = (lane >> 4) << 3;
  BF16x16 u;
  u.q[0] = *(const uint4*)(As_kk + m * LDA + kb);
  u.q[1] = *(const uint4*)(As_kk + m * LDA + 16 + kb);
  return u.v;
}

__device__ __forceinline__ v16bf load_b_frag(const __bf16* BsT_kk, int lane) {
  const int n  = lane & 15;
  const int kb = (lane >> 4) << 3;
  BF16x16 u;
  u.q[0] = *(const uint4*)(BsT_kk + n * LDKB + kb);
  u.q[1] = *(const uint4*)(BsT_kk + n * LDKB + 16 + kb);
  return u.v;
}

__device__ __forceinline__ v8f wmma_bf16(v16bf a, v16bf b, v8f c) {
  return __builtin_amdgcn_wmma_f32_16x16x32_bf16(false, a, false, b,
                                                 (short)0, c, false, false);
}

__device__ __forceinline__ float silu_f(float g) {
  return g / (1.f + __expf(-g));
}

// ---------------------------------------------------------------------------
// Register-staged tile fetch (global -> regs) and stage (regs -> LDS)
// ---------------------------------------------------------------------------
struct FragA  { uint4  q[4]; };   // 32 bf16 per thread
struct FragB4 { float4 v[4]; };   // 16 f32  per thread
struct FragB8 { float4 v[8]; };   // 32 f32  per thread (swiglu dual tile)

__device__ __forceinline__ FragA fetch_a_gather(const __bf16* Ag, const int* idxl,
                                                int cnt, int t0, int k0, int ldaG,
                                                int tid) {
  const int row = tid >> 1;
  const int seg = (tid & 1) * 32;
  int r = t0 + row; if (r >= cnt) r = cnt - 1;     // clamp: results discarded later
  const int tok = idxl[r];
  const uint4* gp = (const uint4*)(Ag + (size_t)tok * ldaG + k0 + seg);
  FragA f;
#pragma unroll
  for (int i = 0; i < 4; ++i) f.q[i] = gp[i];
  return f;
}

__device__ __forceinline__ FragA fetch_a_direct(const __bf16* Ag, int t0, int k0,
                                                int ldaG, int tid) {
  const int row = tid >> 1;
  const int seg = (tid & 1) * 32;
  const uint4* gp = (const uint4*)(Ag + (size_t)(t0 + row) * ldaG + k0 + seg);
  FragA f;
#pragma unroll
  for (int i = 0; i < 4; ++i) f.q[i] = gp[i];
  return f;
}

__device__ __forceinline__ void stage_a(__bf16* As, const FragA& f, int tid) {
  const int row = tid >> 1;
  const int seg = (tid & 1) * 32;
  uint4* lp = (uint4*)(&As[row * LDA + seg]);
#pragma unroll
  for (int i = 0; i < 4; ++i) lp[i] = f.q[i];
}

// NT B tile (B[k][n], n contiguous): each thread fetches a K-row *pair* so the
// stage step packs (k,k+1) into one dword -> one v_cvt_pk + one ds_store_b32
__device__ __forceinline__ FragB4 fetch_b_nt(const float* Bg, int k0, int n0,
                                             int ldbG, int tid) {
  const int kp   = tid >> 3;          // k pair: rows 2kp, 2kp+1 (0..31)
  const int nseg = (tid & 7) * 8;     // 8 columns
  const float* r0 = Bg + (size_t)(k0 + 2 * kp) * ldbG + n0 + nseg;
  const float* r1 = r0 + ldbG;
  FragB4 f;
  f.v[0] = ((const float4*)r0)[0];
  f.v[1] = ((const float4*)r0)[1];
  f.v[2] = ((const float4*)r1)[0];
  f.v[3] = ((const float4*)r1)[1];
  return f;
}

__device__ __forceinline__ void stage_b_nt(__bf16* BsT, const FragB4& f, int tid) {
  const int kp   = tid >> 3;
  const int nseg = (tid & 7) * 8;
#pragma unroll
  for (int i = 0; i < 2; ++i)
#pragma unroll
    for (int j = 0; j < 4; ++j) {
      const int n = nseg + i * 4 + j;
      *(unsigned int*)(&BsT[n * LDKB + 2 * kp]) =
          pk_bf16(f4_at(f.v[i], j), f4_at(f.v[2 + i], j));
    }
}

// BT B tile (B[n][k], k contiguous: down_proj): contiguous reads and pair-packed
// contiguous ds_store_b32 writes
__device__ __forceinline__ FragB4 fetch_b_bt(const float* Bg, int k0, int n0,
                                             int ldbG, int tid) {
  const int n    = tid >> 2;
  const int kseg = (tid & 3) * 16;
  const float4* gp = (const float4*)(Bg + (size_t)(n0 + n) * ldbG + k0 + kseg);
  FragB4 f;
#pragma unroll
  for (int i = 0; i < 4; ++i) f.v[i] = gp[i];
  return f;
}

__device__ __forceinline__ void stage_b_bt(__bf16* BsT, const FragB4& f, int tid) {
  const int n    = tid >> 2;
  const int kseg = (tid & 3) * 16;
#pragma unroll
  for (int i = 0; i < 4; ++i) {
    const float4 v = f.v[i];
    *(unsigned int*)(&BsT[n * LDKB + kseg + i * 4])     = pk_bf16(v.x, v.y);
    *(unsigned int*)(&BsT[n * LDKB + kseg + i * 4 + 2]) = pk_bf16(v.z, v.w);
  }
}

// SwiGLU dual B tile: gate cols [h0,h0+64) and up cols [DIM+h0,DIM+h0+64),
// fetched as K-row pairs for packed staging
__device__ __forceinline__ FragB8 fetch_b_swiglu(const float* gu, int k0, int h0,
                                                 int tid) {
  const int kp   = tid >> 3;          // rows 2kp, 2kp+1
  const int cseg = (tid & 7) * 16;    // 16 local cols, wholly gate or wholly up
  const size_t gcol = (cseg < 64) ? (size_t)(h0 + cseg)
                                  : (size_t)(DIM + h0 + cseg - 64);
  const float* r0 = gu + (size_t)(k0 + 2 * kp) * DIM2 + gcol;
  const float* r1 = r0 + DIM2;
  FragB8 f;
#pragma unroll
  for (int i = 0; i < 4; ++i) f.v[i]     = ((const float4*)r0)[i];
#pragma unroll
  for (int i = 0; i < 4; ++i) f.v[4 + i] = ((const float4*)r1)[i];
  return f;
}

__device__ __forceinline__ void stage_b_swiglu(__bf16* BsT2, const FragB8& f, int tid) {
  const int kp   = tid >> 3;
  const int cseg = (tid & 7) * 16;
#pragma unroll
  for (int i = 0; i < 4; ++i)
#pragma unroll
    for (int j = 0; j < 4; ++j) {
      const int c = cseg + i * 4 + j;
      *(unsigned int*)(&BsT2[c * LDKB + 2 * kp]) =
          pk_bf16(f4_at(f.v[i], j), f4_at(f.v[4 + i], j));
    }
}

__device__ __forceinline__ void mma_step(const __bf16* As, const __bf16* BsT,
                                         int wm, int wn, int lane, v8f acc[2][2]) {
#pragma unroll
  for (int kk = 0; kk < BK; kk += 32) {
    v16bf a0 = load_a_frag(&As[(wm * 32) * LDA + kk], lane);
    v16bf a1 = load_a_frag(&As[(wm * 32 + 16) * LDA + kk], lane);
    v16bf b0 = load_b_frag(&BsT[(wn * 32) * LDKB + kk], lane);
    v16bf b1 = load_b_frag(&BsT[(wn * 32 + 16) * LDKB + kk], lane);
    acc[0][0] = wmma_bf16(a0, b0, acc[0][0]);
    acc[0][1] = wmma_bf16(a0, b1, acc[0][1]);
    acc[1][0] = wmma_bf16(a1, b0, acc[1][0]);
    acc[1][1] = wmma_bf16(a1, b1, acc[1][1]);
  }
}

// ---------------------------------------------------------------------------
// Router: logits = x @ router_w.T + b ; top-4 ; softmax over top vals
// ---------------------------------------------------------------------------
__global__ __launch_bounds__(32)
void router_kernel(const float* __restrict__ x, const float* __restrict__ rw,
                   const float* __restrict__ rb, float* __restrict__ combine) {
  const int t = blockIdx.x;
  const int lane = threadIdx.x;
  float logits[NEXP];
#pragma unroll 1
  for (int e = 0; e < NEXP; ++e) {
    float p = 0.f;
    for (int d = lane; d < DIM; d += 32)
      p += x[(size_t)t * DIM + d] * rw[(size_t)e * DIM + d];
#pragma unroll
    for (int off = 16; off > 0; off >>= 1) p += __shfl_xor(p, off, 32);
    logits[e] = p + rb[e];
  }
  if (lane == 0) {
    bool used[NEXP] = {};
    float tv[TOPK]; int ti[TOPK];
    for (int k = 0; k < TOPK; ++k) {
      float best = -3.4e38f; int bi = 0;
      for (int e = 0; e < NEXP; ++e)
        if (!used[e] && logits[e] > best) { best = logits[e]; bi = e; }
      used[bi] = true; tv[k] = best; ti[k] = bi;
    }
    float m = tv[0], s = 0.f, w[TOPK];
    for (int k = 0; k < TOPK; ++k) { w[k] = __expf(tv[k] - m); s += w[k]; }
    for (int e = 0; e < NEXP; ++e) combine[t * NEXP + e] = 0.f;
    for (int k = 0; k < TOPK; ++k) combine[t * NEXP + ti[k]] = w[k] / s;
  }
}

// Deterministic per-expert token lists (order = ascending token id)
__global__ __launch_bounds__(32)
void build_lists_kernel(const float* __restrict__ combine,
                        int* __restrict__ idx, int* __restrict__ counts) {
  const int e = blockIdx.x;
  if (threadIdx.x != 0) return;
  int c = 0;
  for (int t = 0; t < T_TOK; ++t)
    if (combine[t * NEXP + e] != 0.f) idx[e * T_TOK + c++] = t;
  counts[e] = c;
}

__global__ void cvt_kernel(const float* __restrict__ x, __bf16* __restrict__ xb) {
  const int i = blockIdx.x * blockDim.x + threadIdx.x;
  const float4 v = ((const float4*)x)[i];
  uint2 p;
  p.x = pk_bf16(v.x, v.y);
  p.y = pk_bf16(v.z, v.w);
  *(uint2*)(xb + (size_t)i * 4) = p;
}

__global__ void zero_kernel(float* __restrict__ out) {
  out[(size_t)blockIdx.x * blockDim.x + threadIdx.x] = 0.f;
}

// ---------------------------------------------------------------------------
// SwiGLU GEMM on gathered tokens: inter[r,h] = silu(X@gate)*(X@up), r = list pos
// ---------------------------------------------------------------------------
__global__ __launch_bounds__(256)
void swiglu_gemm_kernel(const __bf16* __restrict__ xb,
                        const float* __restrict__ gu_e,
                        __bf16* __restrict__ inter,
                        const int* __restrict__ idxl,
                        const int* __restrict__ cntp) {
  const int cnt = *cntp;
  const int t0 = blockIdx.y * BM;
  if (t0 >= cnt) return;                       // uniform early-exit

  __shared__ alignas(16) __bf16 As[BM * LDA];
  __shared__ alignas(16) __bf16 BsT[2 * BN * LDKB];
  const int tid = threadIdx.x;
  const int n0 = blockIdx.x * BN;
  const int wave = tid >> 5, lane = tid & 31;
  const int wm = wave >> 1, wn = wave & 1;

  v8f ag[2][2] = {}; v8f au[2][2] = {};

  FragA  fa = fetch_a_gather(xb, idxl, cnt, t0, 0, DIM, tid);
  FragB8 fb = fetch_b_swiglu(gu_e, 0, n0, tid);

  for (int k0 = 0; k0 < DIM; k0 += BK) {
    stage_a(As, fa, tid);
    stage_b_swiglu(BsT, fb, tid);
    __syncthreads();
    if (k0 + BK < DIM) {                       // prefetch next tile into regs
      fa = fetch_a_gather(xb, idxl, cnt, t0, k0 + BK, DIM, tid);
      fb = fetch_b_swiglu(gu_e, k0 + BK, n0, tid);
    }
#pragma unroll
    for (int kk = 0; kk < BK; kk += 32) {
      v16bf a0  = load_a_frag(&As[(wm * 32) * LDA + kk], lane);
      v16bf a1  = load_a_frag(&As[(wm * 32 + 16) * LDA + kk], lane);
      v16bf bg0 = load_b_frag(&BsT[(wn * 32) * LDKB + kk], lane);
      v16bf bg1 = load_b_frag(&BsT[(wn * 32 + 16) * LDKB + kk], lane);
      v16bf bu0 = load_b_frag(&BsT[(64 + wn * 32) * LDKB + kk], lane);
      v16bf bu1 = load_b_frag(&BsT[(64 + wn * 32 + 16) * LDKB + kk], lane);
      ag[0][0] = wmma_bf16(a0, bg0, ag[0][0]);
      ag[0][1] = wmma_bf16(a0, bg1, ag[0][1]);
      ag[1][0] = wmma_bf16(a1, bg0, ag[1][0]);
      ag[1][1] = wmma_bf16(a1, bg1, ag[1][1]);
      au[0][0] = wmma_bf16(a0, bu0, au[0][0]);
      au[0][1] = wmma_bf16(a0, bu1, au[0][1]);
      au[1][0] = wmma_bf16(a1, bu0, au[1][0]);
      au[1][1] = wmma_bf16(a1, bu1, au[1][1]);
    }
    __syncthreads();
  }

  const int rbase = (lane >> 4) << 3;
  const int nlane = lane & 15;
#pragma unroll
  for (int ms = 0; ms < 2; ++ms)
#pragma unroll
    for (int ns = 0; ns < 2; ++ns)
#pragma unroll
      for (int r = 0; r < 8; ++r) {
        const int lr = t0 + wm * 32 + ms * 16 + rbase + r;   // list position
        const int h  = n0 + wn * 32 + ns * 16 + nlane;
        const float s = silu_f(ag[ms][ns][r]) * au[ms][ns][r];
        inter[(size_t)lr * DIM + h] = (__bf16)s;
      }
}

// ---------------------------------------------------------------------------
// comp stage 1 on gathered tokens: hbuf[r,i] = silu(X @ w1[e])
// ---------------------------------------------------------------------------
__global__ __launch_bounds__(256)
void gemm_silu_kernel(const __bf16* __restrict__ xb,
                      const float* __restrict__ w1_e,
                      __bf16* __restrict__ hbuf,
                      const int* __restrict__ idxl,
                      const int* __restrict__ cntp) {
  const int cnt = *cntp;
  const int t0 = blockIdx.y * BM;
  if (t0 >= cnt) return;

  __shared__ alignas(16) __bf16 As[BM * LDA];
  __shared__ alignas(16) __bf16 BsT[BN * LDKB];
  const int tid = threadIdx.x;
  const int n0 = blockIdx.x * BN;
  const int wave = tid >> 5, lane = tid & 31;
  const int wm = wave >> 1, wn = wave & 1;

  v8f acc[2][2] = {};
  FragA  fa = fetch_a_gather(xb, idxl, cnt, t0, 0, DIM, tid);
  FragB4 fb = fetch_b_nt(w1_e, 0, n0, ICOMP, tid);

  for (int k0 = 0; k0 < DIM; k0 += BK) {
    stage_a(As, fa, tid);
    stage_b_nt(BsT, fb, tid);
    __syncthreads();
    if (k0 + BK < DIM) {
      fa = fetch_a_gather(xb, idxl, cnt, t0, k0 + BK, DIM, tid);
      fb = fetch_b_nt(w1_e, k0 + BK, n0, ICOMP, tid);
    }
    mma_step(As, BsT, wm, wn, lane, acc);
    __syncthreads();
  }

  const int rbase = (lane >> 4) << 3;
  const int nlane = lane & 15;
#pragma unroll
  for (int ms = 0; ms < 2; ++ms)
#pragma unroll
    for (int ns = 0; ns < 2; ++ns)
#pragma unroll
      for (int r = 0; r < 8; ++r) {
        const int lr = t0 + wm * 32 + ms * 16 + rbase + r;
        const int n  = n0 + wn * 32 + ns * 16 + nlane;
        hbuf[(size_t)lr * ICOMP + n] = (__bf16)silu_f(acc[ms][ns][r]);
      }
}

// ---------------------------------------------------------------------------
// Scaled scatter GEMM: out[tok,n] += combine[tok,eidx] * (A @ B)[r,n]
// A indexed by list position (compacted); tok = idxl[r].
// BT=true: B[n][k] (down_proj); BT=false: B[k][n] (comp_w2)
// ---------------------------------------------------------------------------
template <bool BT>
__global__ __launch_bounds__(256)
void gemm_scaled_scatter_kernel(const __bf16* __restrict__ Ab,
                                const float* __restrict__ Bg,
                                const float* __restrict__ combine,
                                float* __restrict__ out,
                                const int* __restrict__ idxl,
                                const int* __restrict__ cntp,
                                int ldaG, int ldbG, int Ktot, int eidx) {
  const int cnt = *cntp;
  const int t0 = blockIdx.y * BM;
  if (t0 >= cnt) return;

  __shared__ alignas(16) __bf16 As[BM * LDA];
  __shared__ alignas(16) __bf16 BsT[BN * LDKB];
  const int tid = threadIdx.x;
  const int n0 = blockIdx.x * BN;
  const int wave = tid >> 5, lane = tid & 31;
  const int wm = wave >> 1, wn = wave & 1;

  v8f acc[2][2] = {};
  FragA  fa = fetch_a_direct(Ab, t0, 0, ldaG, tid);
  FragB4 fb = BT ? fetch_b_bt(Bg, 0, n0, ldbG, tid)
                 : fetch_b_nt(Bg, 0, n0, ldbG, tid);

  for (int k0 = 0; k0 < Ktot; k0 += BK) {
    stage_a(As, fa, tid);
    if (BT) stage_b_bt(BsT, fb, tid); else stage_b_nt(BsT, fb, tid);
    __syncthreads();
    if (k0 + BK < Ktot) {
      fa = fetch_a_direct(Ab, t0, k0 + BK, ldaG, tid);
      fb = BT ? fetch_b_bt(Bg, k0 + BK, n0, ldbG, tid)
              : fetch_b_nt(Bg, k0 + BK, n0, ldbG, tid);
    }
    mma_step(As, BsT, wm, wn, lane, acc);
    __syncthreads();
  }

  const int rbase = (lane >> 4) << 3;
  const int nlane = lane & 15;
#pragma unroll
  for (int ms = 0; ms < 2; ++ms) {
#pragma unroll
    for (int r = 0; r < 8; ++r) {
      const int lr = t0 + wm * 32 + ms * 16 + rbase + r;
      if (lr < cnt) {
        const int tok = idxl[lr];
        const float sc = combine[tok * NEXP + eidx];
#pragma unroll
        for (int ns = 0; ns < 2; ++ns) {
          const int n = n0 + wn * 32 + ns * 16 + nlane;
          atomicAdd(&out[(size_t)tok * DIM + n], sc * acc[ms][ns][r]);
        }
      }
    }
  }
}

// ---------------------------------------------------------------------------
extern "C" void kernel_launch(void* const* d_in, const int* in_sizes, int n_in,
                              void* d_out, int out_size, void* d_ws, size_t ws_size,
                              hipStream_t stream) {
  (void)in_sizes; (void)n_in; (void)out_size; (void)ws_size;
  const float* x   = (const float*)d_in[0];
  const float* rw  = (const float*)d_in[1];
  const float* rb  = (const float*)d_in[2];
  const float* gup = (const float*)d_in[3];
  const float* dwn = (const float*)d_in[4];
  const float* w1  = (const float*)d_in[5];
  const float* w2  = (const float*)d_in[6];
  float* out = (float*)d_out;
  char* ws = (char*)d_ws;

  float*  combine = (float*)ws;                                     // 48 KB
  int*    idx     = (int*)(ws + (1u << 16));                        // 48 KB
  int*    counts  = (int*)(ws + (2u << 16));                        // 48 B
  __bf16* xb      = (__bf16*)(ws + (3u << 16));                     // 5.9 MB
  __bf16* inter   = (__bf16*)(ws + (3u << 16) + 6u * 1024 * 1024);  // 5.9 MB
  __bf16* hbuf    = (__bf16*)(ws + (3u << 16) + 12u * 1024 * 1024); // 1 MB

  zero_kernel<<<(T_TOK * DIM) / 256, 256, 0, stream>>>(out);
  router_kernel<<<T_TOK, 32, 0, stream>>>(x, rw, rb, combine);
  build_lists_kernel<<<NEXP, 32, 0, stream>>>(combine, idx, counts);
  cvt_kernel<<<(T_TOK * DIM) / (4 * 256), 256, 0, stream>>>(x, xb);

  for (int e = 0; e < NN; ++e) {
    swiglu_gemm_kernel<<<dim3(DIM / BN, T_TOK / BM), 256, 0, stream>>>(
        xb, gup + (size_t)e * DIM * DIM2, inter, idx + e * T_TOK, counts + e);
    gemm_scaled_scatter_kernel<true><<<dim3(DIM / BN, T_TOK / BM), 256, 0, stream>>>(
        inter, dwn + (size_t)e * DIM * DIM, combine, out,
        idx + e * T_TOK, counts + e, DIM, DIM, DIM, e);
  }
  for (int e = 0; e < NC; ++e) {
    const int es = NN + e;
    gemm_silu_kernel<<<dim3(ICOMP / BN, T_TOK / BM), 256, 0, stream>>>(
        xb, w1 + (size_t)e * DIM * ICOMP, hbuf, idx + es * T_TOK, counts + es);
    gemm_scaled_scatter_kernel<false><<<dim3(DIM / BN, T_TOK / BM), 256, 0, stream>>>(
        hbuf, w2 + (size_t)e * ICOMP * DIM, combine, out,
        idx + es * T_TOK, counts + es, ICOMP, DIM, ICOMP, es);
  }
}
